// SoftNCutsLoss_v1_38611755991338
// MI455X (gfx1250) — matrix operand order repaired and checked by reading, not compiled, not run on hardware
//
#include <hip/hip_runtime.h>
#include <hip/hip_fp16.h>

typedef __attribute__((ext_vector_type(16))) _Float16 v16h;
typedef __attribute__((ext_vector_type(8)))  float    v8f;

#define N_TOT 8192   // 8*32*32
#define K_CLS 8

// ---------------- init: zero the num/den accumulators ----------------
__global__ void sncl_init(float* acc) {
  int t = threadIdx.x;
  if (t < 16) acc[t] = 0.0f;
}

// ---------------- prep: build transposed, augmented f16 B ----------------
// Bt[c][j], c=0..7 classes, c=8 ones (gives W row-sums), c=9..15 zero pad.
__global__ void sncl_prep(const float* __restrict__ prob,
                          _Float16* __restrict__ Bt) {
  int j = blockIdx.x * blockDim.x + threadIdx.x;
  if (j >= N_TOT) return;
#pragma unroll
  for (int c = 0; c < 8; ++c)
    Bt[c * N_TOT + j] = (_Float16)prob[j * 8 + c];
  Bt[8 * N_TOT + j] = (_Float16)1.0f;
#pragma unroll
  for (int c = 9; c < 16; ++c)
    Bt[c * N_TOT + j] = (_Float16)0.0f;
}

// ---------------- main: fused W-generation + WMMA GEMM ----------------
__global__ __launch_bounds__(256) void sncl_main(
    const float* __restrict__ patch, const float* __restrict__ prob,
    const _Float16* __restrict__ Bt, float* __restrict__ accum) {
  __shared__ float fS[N_TOT];      // 32 KB: intensities
  __shared__ float scr[8 * 256];   // 8 KB: per-wave partial C tiles

  const int tid  = threadIdx.x;
  const int wave = tid >> 5;
  const int lane = tid & 31;

  // stage intensities to LDS (hot, scattered reuse below)
  for (int j = tid; j < N_TOT; j += 256) fS[j] = patch[j];
  __syncthreads();

  const int   iBase = blockIdx.x * 16;
  const int   M     = lane & 15;          // A-matrix row owned by this lane
  const int   i     = iBase + M;
  const float fi    = fS[i];
  const int   xi = i >> 10, yi = (i >> 5) & 31, zi = i & 31;
  const int   kOff  = (lane & 16) >> 1;   // lanes 0-15 -> K base 0, 16-31 -> 16 (kOff << 1)

  v8f c = {};

  const int jStart = wave * 1024;         // each wave sweeps 1024 j's
  const _Float16* btLane = Bt + (size_t)(lane & 15) * N_TOT + (lane & 16);

  for (int j0 = jStart; j0 < jStart + 1024; j0 += 32) {
    // ---- build A = 16x32 f16 tile of W, one exp per element ----
    v16h a;
#pragma unroll
    for (int r = 0; r < 8; ++r) {
      const int kk = (kOff << 1) + (r << 1);
#pragma unroll
      for (int s = 0; s < 2; ++s) {
        const int j  = j0 + kk + s;
        const float fj = fS[j];
        const float df = (fi - fj) * (1.0f / 3.0f);   // std_intensity = 3
        const int xj = j >> 10, yj = (j >> 5) & 31, zj = j & 31;
        const int dx = xi - xj, dy = yi - yj, dz = zi - zj;
        const float e = -(df * df + (float)(dx * dx + dy * dy + dz * dz));
        a[2 * r + s] = (_Float16)__expf(e);           // merged Gaussians
      }
    }
    // ---- B = 32x16 f16 tile of [P | 1 | 0...] : 32 contiguous bytes/lane ----
    const v16h b = *(const v16h*)(btLane + j0);
    // ---- D = A*B + C  (v_wmma_f32_16x16x32_f16) ----
    c = __builtin_amdgcn_wmma_f32_16x16x32_f16(false, a, false, b,
                                               (short)0, c, false, false);
  }

  // ---- reduce the 8 per-wave partial C tiles through LDS ----
  *(v8f*)&scr[wave * 256 + lane * 8] = c;
  __syncthreads();

  if (tid < 128) {
    const int Mr = tid >> 3, Nc = tid & 7;
    const int r       = Mr & 7;                      // C VGPR index
    const int laneSel = ((Mr & 8) ? 16 : 0) + Nc;    // lane holding (Mr,Nc)
    const int laneRow = ((Mr & 8) ? 16 : 0) + 8;     // lane holding (Mr, col8=rowsum)
    float wp = 0.f, rs = 0.f;
#pragma unroll
    for (int w = 0; w < 8; ++w) {
      wp += scr[w * 256 + laneSel * 8 + r];
      rs += scr[w * 256 + laneRow * 8 + r];
    }
    const float p = prob[(size_t)(iBase + Mr) * 8 + Nc];
    atomicAdd(&accum[Nc],     p * wp);   // num_c partial
    atomicAdd(&accum[8 + Nc], p * rs);   // den_c partial
  }
}

// ---------------- finish: loss = k - sum(num/den) ----------------
__global__ void sncl_finish(const float* __restrict__ accum,
                            float* __restrict__ out) {
  if (threadIdx.x == 0) {
    float s = 0.f;
#pragma unroll
    for (int c2 = 0; c2 < 8; ++c2) s += accum[c2] / accum[8 + c2];
    out[0] = 8.0f - s;
  }
}

extern "C" void kernel_launch(void* const* d_in, const int* in_sizes, int n_in,
                              void* d_out, int out_size, void* d_ws, size_t ws_size,
                              hipStream_t stream) {
  const float* patch = (const float*)d_in[0];   // [8,32,32] f32
  const float* prob  = (const float*)d_in[1];   // [8,32,32,8] f32
  // d_in[2] = k (==8), fixed at compile time
  float*     out   = (float*)d_out;
  float*     accum = (float*)d_ws;                         // 16 floats
  _Float16*  Bt    = (_Float16*)((char*)d_ws + 256);       // 16*8192*2 = 256 KB

  sncl_init  <<<1, 32, 0, stream>>>(accum);
  sncl_prep  <<<N_TOT / 256, 256, 0, stream>>>(prob, Bt);
  sncl_main  <<<N_TOT / 16, 256, 0, stream>>>(patch, prob, Bt, accum);
  sncl_finish<<<1, 32, 0, stream>>>(accum, out);
}
